// AttentionLSTMLanguageModel_59785944760578
// MI455X (gfx1250) — compile-verified, matrix operand order checked by reading, not run on hardware
//
#include <hip/hip_runtime.h>
#include <hip/hip_bf16.h>
#include <math.h>

// ---------------------------------------------------------------------------
// Types for CDNA5 WMMA (wave32): 16x16x32 bf16 -> f32
// ---------------------------------------------------------------------------
typedef __bf16 bf16_t;
typedef __attribute__((ext_vector_type(16))) __bf16 v16bf;
typedef __attribute__((ext_vector_type(8)))  float  v8f;

union FragU { uint4 q[2]; v16bf v; };

__device__ __forceinline__ v8f vzero() {
  v8f v;
#pragma unroll
  for (int i = 0; i < 8; ++i) v[i] = 0.0f;
  return v;
}

// Load a 16x32 bf16 operand tile (A-layout; also used for B as rows of [N,K]).
// ISA 16-bit A-matrix 16x32 layout (CDNA5 ISA):
//   lanes 0-15 : row = lane,     K = k0+0..7   (elems 0-7), k0+16..23 (elems 8-15)
//   lanes16-31 : row = lane-16,  K = k0+8..15  (elems 0-7), k0+24..31 (elems 8-15)
__device__ __forceinline__ v16bf load_frag(const bf16_t* base, int ld, int row,
                                           int k0, int lane) {
  int r, ka;
  if (lane < 16) { r = row + lane;      ka = k0;     }
  else           { r = row + lane - 16; ka = k0 + 8; }
  const bf16_t* p = base + (size_t)r * ld + ka;
  FragU u;
  u.q[0] = *(const uint4*)(p);       // 8 bf16: ka .. ka+7
  u.q[1] = *(const uint4*)(p + 16);  // 8 bf16: ka+16 .. ka+23
  return u.v;
}

// Same fragment swizzle, but from an LDS-staged 32-wide K-tile with row
// stride 40 bf16 (80B) to avoid ds_load_b128 bank conflicts.
__device__ __forceinline__ v16bf load_frag_lds(const bf16_t* base, int row,
                                               int lane) {
  int r  = row + (lane & 15);
  int ka = (lane < 16) ? 0 : 8;
  const bf16_t* p = base + r * 40 + ka;
  FragU u;
  u.q[0] = *(const uint4*)(p);
  u.q[1] = *(const uint4*)(p + 16);
  return u.v;
}

// CDNA5 async global->LDS copy of one 16B chunk (ASYNCcnt-tracked).
__device__ __forceinline__ void async_copy16(unsigned lds_off, const void* gptr) {
  asm volatile("global_load_async_to_lds_b128 %0, %1, off"
               :: "v"(lds_off), "v"((unsigned long long)(uintptr_t)gptr)
               : "memory");
}
__device__ __forceinline__ void wait_async0() {
  asm volatile("s_wait_asynccnt 0x0" ::: "memory");
}

__device__ __forceinline__ float sigmoid_f(float x) {
  return 1.0f / (1.0f + expf(-x));
}

// ---------------------------------------------------------------------------
// Small prep kernels
// ---------------------------------------------------------------------------
__global__ void cvt_bf16_kernel(const float* __restrict__ in,
                                bf16_t* __restrict__ out, int n) {
  int i = blockIdx.x * blockDim.x + threadIdx.x;
  if (i < n) out[i] = (bf16_t)in[i];
}

__global__ void bias_sum_kernel(const float* __restrict__ a,
                                const float* __restrict__ b,
                                float* __restrict__ out, int n) {
  int i = blockIdx.x * blockDim.x + threadIdx.x;
  if (i < n) out[i] = a[i] + b[i];
}

// x_bf16[r][d] = bf16(emb_W[src[r]][d]);  r = b*T + t, d in [0,512)
__global__ void embed_kernel(const int* __restrict__ src,
                             const float* __restrict__ embW,
                             bf16_t* __restrict__ xB, int n) {
  int i = blockIdx.x * blockDim.x + threadIdx.x;
  if (i < n) {
    int r = i >> 9;
    int d = i & 511;
    int tok = src[r];
    xB[i] = (bf16_t)embW[(size_t)tok * 512 + d];
  }
}

// ---------------------------------------------------------------------------
// Generic bf16 WMMA GEMM:  C[M,N] = A[M,K] @ B[N,K]^T + bias[N], opt tanh.
// Block = 256 threads (8 waves); workgroup tile 32x128; wave tile 16x32.
// Operands loaded straight from global (L2-resident). Used for the small and
// medium GEMMs.
// ---------------------------------------------------------------------------
__global__ __launch_bounds__(256) void gemm_bf16_kernel(
    const bf16_t* __restrict__ A, const bf16_t* __restrict__ Bw,
    const float* __restrict__ bias, float* __restrict__ Cf,
    bf16_t* __restrict__ Cb, int M, int N, int K, int act) {
  const int lane = threadIdx.x & 31;
  const int w    = threadIdx.x >> 5;
  const int m0 = blockIdx.y * 32 + (w >> 2) * 16;
  const int n0 = blockIdx.x * 128 + (w & 3) * 32;

  v8f acc0 = vzero();
  v8f acc1 = vzero();
  for (int k0 = 0; k0 < K; k0 += 32) {
    v16bf a  = load_frag(A,  K, m0,      k0, lane);
    v16bf b0 = load_frag(Bw, K, n0,      k0, lane);
    v16bf b1 = load_frag(Bw, K, n0 + 16, k0, lane);
    acc0 = __builtin_amdgcn_wmma_f32_16x16x32_bf16(false, a, false, b0,
                                                   (short)0, acc0, false, false);
    acc1 = __builtin_amdgcn_wmma_f32_16x16x32_bf16(false, a, false, b1,
                                                   (short)0, acc1, false, false);
  }

  // C/D layout: VGPR v, lanes 0-15 -> (M=m0+v, N=lane); lanes 16-31 -> (M=m0+8+v)
  const int row_off = (lane >> 4) * 8;
  const int cl = lane & 15;
#pragma unroll
  for (int v = 0; v < 8; ++v) {
    int row = m0 + row_off + v;
    int c0 = n0 + cl;
    int c1 = n0 + 16 + cl;
    float x0 = acc0[v] + bias[c0];
    float x1 = acc1[v] + bias[c1];
    if (act == 1) { x0 = tanhf(x0); x1 = tanhf(x1); }
    size_t o0 = (size_t)row * N + c0;
    size_t o1 = (size_t)row * N + c1;
    if (Cf) { Cf[o0] = x0;          Cf[o1] = x1; }
    if (Cb) { Cb[o0] = (bf16_t)x0;  Cb[o1] = (bf16_t)x1; }
  }
}

// ---------------------------------------------------------------------------
// Decoder GEMM with CDNA5 async global->LDS staging of the B tile.
// C[M,N] = A[M,K] @ B[N,K]^T + bias[N].  Block = 256 threads (8 waves),
// workgroup tile 32x256, wave tile 16x64 (4 accumulators, 1 shared A-frag).
// Per K-step the 16KB B tile (256 rows x 32 K) is copied into LDS with
// global_load_async_to_lds_b128 (double-buffered, ASYNCcnt-tracked) while
// WMMAs consume the previous buffer; fragments are then read via
// ds_load_b128 from conflict-free 80B-padded rows.
// ---------------------------------------------------------------------------
__global__ __launch_bounds__(256) void gemm_bf16_async_kernel(
    const bf16_t* __restrict__ A, const bf16_t* __restrict__ Bw,
    const float* __restrict__ bias, float* __restrict__ Cf,
    int M, int N, int K, int act) {
  __shared__ __align__(16) bf16_t Bs[2][256 * 40];  // 2 x 20KB

  const int tid  = threadIdx.x;
  const int lane = tid & 31;
  const int w    = tid >> 5;
  const int m0   = blockIdx.y * 32 + (w >> 2) * 16;
  const int nblk = blockIdx.x * 256;
  const int ncol = (w & 3) * 64;  // wave's 64-column slice within the tile

  // issue async copies of the 32-wide K-slab starting at k0 into Bs[buf]
  auto issue = [&](int buf, int k0) {
#pragma unroll
    for (int j = 0; j < 4; ++j) {
      int c   = tid + 256 * j;     // 1024 chunks of 16B
      int r   = c >> 2;            // B row within tile
      int seg = c & 3;             // 16B segment within the 64B row-slab
      unsigned lds_off =
          (unsigned)(uintptr_t)&Bs[buf][r * 40 + seg * 8];
      const void* g = Bw + (size_t)(nblk + r) * K + k0 + seg * 8;
      async_copy16(lds_off, g);
    }
  };

  v8f acc[4];
#pragma unroll
  for (int s = 0; s < 4; ++s) acc[s] = vzero();

  issue(0, 0);
  wait_async0();
  __syncthreads();

  for (int k0 = 0; k0 < K; k0 += 32) {
    const int cur = (k0 >> 5) & 1;
    if (k0 + 32 < K) issue(1 - cur, k0 + 32);

    v16bf a = load_frag(A, K, m0, k0, lane);  // A direct from global/L2
#pragma unroll
    for (int s = 0; s < 4; ++s) {
      v16bf b = load_frag_lds(&Bs[cur][0], ncol + s * 16, lane);
      acc[s] = __builtin_amdgcn_wmma_f32_16x16x32_bf16(
          false, a, false, b, (short)0, acc[s], false, false);
    }

    wait_async0();    // next buffer's copies have landed (this wave)
    __syncthreads();  // all waves: done reading cur, next buffer visible
  }

  const int row_off = (lane >> 4) * 8;
  const int cl = lane & 15;
#pragma unroll
  for (int s = 0; s < 4; ++s) {
    int c = nblk + ncol + s * 16 + cl;
    float bv = bias[c];
#pragma unroll
    for (int v = 0; v < 8; ++v) {
      int row = m0 + row_off + v;
      float x = acc[s][v] + bv;
      if (act == 1) x = tanhf(x);
      Cf[(size_t)row * N + c] = x;
    }
  }
}

// ---------------------------------------------------------------------------
// Persistent single-workgroup LSTM over T=512 steps.
// 1024 threads = 32 waves; per step each wave computes 4 of the 128 gate
// N-tiles of g = h @ W_hh^T (+ precomputed Xproj) via bf16 WMMA, then all
// threads do the elementwise cell update. h kept in LDS (bf16, M padded to
// 16), c kept in registers. Intra-WGP s_barrier per phase.
// LDS: 16*512*2 (h) + 8*2048*2 (gates f16) = 48KB.
// ---------------------------------------------------------------------------
__global__ __launch_bounds__(1024) void lstm_kernel(
    const float* __restrict__ Xproj,   // [4096][2048], row = b*512 + t
    const bf16_t* __restrict__ Whh,    // [2048][512] bf16
    float* __restrict__ hs,            // [4096][512] f32 out
    bf16_t* __restrict__ hbf) {        // [4096][512] bf16 out
  __shared__ __align__(16) bf16_t   h_s[16 * 512];
  __shared__ __align__(16) _Float16 g_s[8 * 2048];

  const int tid  = threadIdx.x;
  const int lane = tid & 31;
  const int w    = tid >> 5;

  for (int i = tid; i < 16 * 512; i += 1024) h_s[i] = (bf16_t)0.0f;
  float c_reg[4] = {0.f, 0.f, 0.f, 0.f};
  __syncthreads();

  for (int t = 0; t < 512; ++t) {
    // ---- phase A: gates = h @ Whh^T (M=16 padded, N=64 per wave, K=512)
    v8f acc[4];
#pragma unroll
    for (int j = 0; j < 4; ++j) acc[j] = vzero();
    for (int k0 = 0; k0 < 512; k0 += 32) {
      v16bf a = load_frag(h_s, 512, 0, k0, lane);
#pragma unroll
      for (int j = 0; j < 4; ++j) {
        v16bf b = load_frag(Whh, 512, (w * 4 + j) * 16, k0, lane);
        acc[j] = __builtin_amdgcn_wmma_f32_16x16x32_bf16(
            false, a, false, b, (short)0, acc[j], false, false);
      }
    }
    if (lane < 16) {  // valid batch rows 0..7 live in lanes 0-15, VGPRs 0-7
#pragma unroll
      for (int j = 0; j < 4; ++j) {
        int col = (w * 4 + j) * 16 + lane;
#pragma unroll
        for (int v = 0; v < 8; ++v) {
          float val = acc[j][v] + Xproj[((size_t)v * 512 + t) * 2048 + col];
          g_s[v * 2048 + col] = (_Float16)val;
        }
      }
    }
    __syncthreads();

    // ---- phase B: elementwise LSTM cell; thread owns cells tid + 1024*q
#pragma unroll
    for (int q = 0; q < 4; ++q) {
      int cell = tid + 1024 * q;
      int b = cell >> 9;
      int h = cell & 511;
      float ig = sigmoid_f((float)g_s[b * 2048 + h]);
      float fg = sigmoid_f((float)g_s[b * 2048 + 512 + h]);
      float gg = tanhf((float)g_s[b * 2048 + 1024 + h]);
      float og = sigmoid_f((float)g_s[b * 2048 + 1536 + h]);
      float c = fg * c_reg[q] + ig * gg;
      c_reg[q] = c;
      float hv = og * tanhf(c);
      h_s[b * 512 + h] = (bf16_t)hv;
      size_t r = (size_t)b * 512 + t;
      hs[r * 512 + h]  = hv;
      hbf[r * 512 + h] = (bf16_t)hv;
    }
    __syncthreads();
  }
}

// ---------------------------------------------------------------------------
// s[r] = dot(tanhP[r,:], attn2_W) + attn2_b  (one wave per row)
// ---------------------------------------------------------------------------
__global__ __launch_bounds__(256) void attn_score_kernel(
    const float* __restrict__ P, const float* __restrict__ w2,
    const float* __restrict__ b2, float* __restrict__ s) {
  const int lane = threadIdx.x & 31;
  const int wid  = threadIdx.x >> 5;
  const int r = blockIdx.x * 8 + wid;
  float acc = 0.0f;
  for (int h = lane; h < 512; h += 32) acc += P[(size_t)r * 512 + h] * w2[h];
#pragma unroll
  for (int off = 16; off > 0; off >>= 1) acc += __shfl_down(acc, off, 32);
  if (lane == 0) s[r] = acc + b2[0];
}

// ---------------------------------------------------------------------------
// Prefix-softmax context + build concat input [ctx | h] as bf16.
// grid = (8 col-chunks, 8 batches); block = 64 threads (one h-column each).
// ---------------------------------------------------------------------------
__global__ __launch_bounds__(64) void ctx_kernel(
    const float* __restrict__ s,   // [8*512]
    const float* __restrict__ hs,  // [4096][512]
    bf16_t* __restrict__ cat) {    // [4096][1024]
  __shared__ float e_s[512];
  __shared__ float den_s[512];
  __shared__ float red[64];
  const int b = blockIdx.y;
  const int tid = threadIdx.x;

  float m = -3.4e38f;
  for (int t = tid; t < 512; t += 64) m = fmaxf(m, s[b * 512 + t]);
  red[tid] = m;
  __syncthreads();
  for (int st = 32; st > 0; st >>= 1) {
    if (tid < st) red[tid] = fmaxf(red[tid], red[tid + st]);
    __syncthreads();
  }
  m = red[0];
  for (int t = tid; t < 512; t += 64) e_s[t] = expf(s[b * 512 + t] - m);
  __syncthreads();
  if (tid == 0) {
    float a = 0.0f;
    for (int t = 0; t < 512; ++t) { a += e_s[t]; den_s[t] = a; }
  }
  __syncthreads();

  const int h = blockIdx.x * 64 + tid;
  float acc = 0.0f;
  for (int t = 0; t < 512; ++t) {
    size_t r = (size_t)b * 512 + t;
    float hv = hs[r * 512 + h];
    acc += e_s[t] * hv;
    float ctx = acc / den_s[t];
    cat[r * 1024 + h]       = (bf16_t)ctx;
    cat[r * 1024 + 512 + h] = (bf16_t)hv;
  }
}

// ---------------------------------------------------------------------------
// Host launcher
// ---------------------------------------------------------------------------
extern "C" void kernel_launch(void* const* d_in, const int* in_sizes, int n_in,
                              void* d_out, int out_size, void* d_ws, size_t ws_size,
                              hipStream_t stream) {
  (void)in_sizes; (void)n_in; (void)out_size; (void)ws_size;
  const int V = 32000, D = 512, H = 512, B = 8, T = 512;
  const int M = B * T;        // 4096 rows
  const int G = 4 * H;        // 2048 gate cols

  const int*   src      = (const int*)  d_in[0];
  const float* emb_W    = (const float*)d_in[1];
  const float* W_ih     = (const float*)d_in[2];
  const float* W_hh     = (const float*)d_in[3];
  const float* b_ih     = (const float*)d_in[4];
  const float* b_hh     = (const float*)d_in[5];
  const float* attn1_W  = (const float*)d_in[6];
  const float* attn1_b  = (const float*)d_in[7];
  const float* attn2_W  = (const float*)d_in[8];
  const float* attn2_b  = (const float*)d_in[9];
  const float* concat_W = (const float*)d_in[10];
  const float* concat_b = (const float*)d_in[11];
  const float* dec_b    = (const float*)d_in[12];
  float* out = (float*)d_out;

  // workspace bump allocator (256B aligned)
  char* p = (char*)d_ws;
  auto alloc = [&](size_t bytes) -> void* {
    void* r = (void*)p;
    p += (bytes + 255) & ~(size_t)255;
    return r;
  };
  bf16_t* embB   = (bf16_t*)alloc((size_t)V * D * 2);
  bf16_t* WihB   = (bf16_t*)alloc((size_t)G * D * 2);
  bf16_t* WhhB   = (bf16_t*)alloc((size_t)G * H * 2);
  bf16_t* a1B    = (bf16_t*)alloc((size_t)H * H * 2);
  bf16_t* ccB    = (bf16_t*)alloc((size_t)H * 2 * H * 2);
  float*  bsum   = (float*) alloc((size_t)G * 4);
  bf16_t* xB     = (bf16_t*)alloc((size_t)M * D * 2);
  float*  Xproj  = (float*) alloc((size_t)M * G * 4);
  float*  hsF    = (float*) alloc((size_t)M * H * 4);
  bf16_t* hB     = (bf16_t*)alloc((size_t)M * H * 2);
  float*  Pf     = (float*) alloc((size_t)M * H * 4);
  float*  sV     = (float*) alloc((size_t)M * 4);
  bf16_t* catB   = (bf16_t*)alloc((size_t)M * 2 * H * 2);
  bf16_t* combB  = (bf16_t*)alloc((size_t)M * H * 2);

  auto cvt = [&](const float* in, bf16_t* o, int n) {
    cvt_bf16_kernel<<<(n + 255) / 256, 256, 0, stream>>>(in, o, n);
  };

  // 1) weight conversions to bf16
  cvt(emb_W,    embB, V * D);
  cvt(W_ih,     WihB, G * D);
  cvt(W_hh,     WhhB, G * H);
  cvt(attn1_W,  a1B,  H * H);
  cvt(concat_W, ccB,  H * 2 * H);
  bias_sum_kernel<<<(G + 255) / 256, 256, 0, stream>>>(b_ih, b_hh, bsum, G);

  // 2) embedding gather -> bf16 activations
  embed_kernel<<<(M * D + 255) / 256, 256, 0, stream>>>(src, emb_W, xB, M * D);

  // 3) Xproj = x @ W_ih^T + (b_ih + b_hh)   [4096 x 2048]
  gemm_bf16_kernel<<<dim3(G / 128, M / 32), 256, 0, stream>>>(
      xB, WihB, bsum, Xproj, nullptr, M, G, D, 0);

  // 4) sequential LSTM recurrence (persistent single workgroup)
  lstm_kernel<<<1, 1024, 0, stream>>>(Xproj, WhhB, hsF, hB);

  // 5) P = tanh(h @ attn1_W^T + attn1_b)    [4096 x 512]
  gemm_bf16_kernel<<<dim3(H / 128, M / 32), 256, 0, stream>>>(
      hB, a1B, attn1_b, Pf, nullptr, M, H, H, 1);

  // 6) attention scores s = P . attn2_W + attn2_b
  attn_score_kernel<<<M / 8, 256, 0, stream>>>(Pf, attn2_W, attn2_b, sV);

  // 7) prefix-softmax context + concat [ctx | h] -> bf16
  ctx_kernel<<<dim3(H / 64, B), 64, 0, stream>>>(sV, hsF, catB);

  // 8) comb = tanh(cat @ concat_W^T + concat_b)   [4096 x 512], bf16 out
  gemm_bf16_kernel<<<dim3(H / 128, M / 32), 256, 0, stream>>>(
      catB, ccB, concat_b, nullptr, combB, M, H, 2 * H, 1);

  // 9) logits = comb @ emb_W^T + dec_b  [4096 x 32000] f32, async-LDS staged
  gemm_bf16_async_kernel<<<dim3(V / 256, M / 32), 256, 0, stream>>>(
      combB, embB, dec_b, out, M, V, D, 0);
}
